// VisionMamba_53755810677239
// MI455X (gfx1250) — matrix operand order repaired and checked by reading, not compiled, not run on hardware
//
#include <hip/hip_runtime.h>
#include <hip/hip_bf16.h>
#include <math.h>
#include <stdint.h>

// ---------------- model constants ----------------
static constexpr int IMG    = 224;
static constexpr int PATCH  = 16;
static constexpr int LSEQ   = 196;
static constexpr int DMODEL = 384;
static constexpr int NDEPTH = 12;
static constexpr int DINNER = 768;
static constexpr int DSTATE = 16;
static constexpr int DCONV  = 4;
static constexpr int DTRANK = 24;
static constexpr int NB     = 32;
static constexpr int ROWS   = NB * LSEQ;        // 6272 (multiple of 64!)
static constexpr int NDEV   = 7;
static constexpr int NDIST  = 4;
static constexpr int DBC_N  = DTRANK + 2 * DSTATE; // 56

typedef __attribute__((ext_vector_type(16))) _Float16 v16h;
typedef __attribute__((ext_vector_type(8)))  _Float16 v8h;
typedef __attribute__((ext_vector_type(4)))  _Float16 v4h;
typedef __attribute__((ext_vector_type(8)))  float    v8f;
typedef __attribute__((ext_vector_type(4)))  float    v4f;

// ---------------- WMMA GEMM: C[M,N] (op) A[M,K] * W, f32 in/out, f16 compute ----
// Assumes M % 64 == 0 (always true here: M == 6272).
// GK: K is ragged (guard k).  GN: N is ragged (guard n).
// wt==0: W is [K,N] row-major (ldw = N).  wt==1: W is [N,K] row-major (ldw = K).
// mode 0: C = v (+bias[n] if bias)     mode 1: C = softplus(v + bias[n])
// mode 2: C += v (residual accumulate) mode 3: C = v + bias[n] + pos[(m%L)*DMODEL+n]
template <bool GK, bool GN>
__global__ __launch_bounds__(256) void gemm_wmma_f16(
    const float* __restrict__ A, int lda,
    const float* __restrict__ W, int ldw, int wt,
    const float* __restrict__ bias,
    const float* __restrict__ pos,
    float* __restrict__ C, int ldc,
    int M, int N, int K, int mode)
{
  // LDS tiles, 40-half row pitch (80B rows: 16B-aligned, conflict-free frags)
  __shared__ __align__(16) _Float16 As[64 * 40];
  __shared__ __align__(16) _Float16 Bs[64 * 40];

  const int tid  = threadIdx.x;
  const int lane = tid & 31;
  const int wave = tid >> 5;
  const int wm   = wave & 3;   // 4 m-tiles of 16 -> BM=64
  const int wn   = wave >> 2;  // 2 n-strips of 32 -> BN=64
  const int bm   = blockIdx.y * 64;
  const int bn   = blockIdx.x * 64;
  const int mrow = bm + wm * 16;
  const int ncol0 = bn + wn * 32;

  v8f acc0 = {};
  v8f acc1 = {};

  for (int k0 = 0; k0 < K; k0 += 32) {
    if (!GK && k0 + 32 < K) {
      // next A K-slab prefetch (global_prefetch_b8); always in-bounds (M%64==0)
      __builtin_prefetch(&A[(size_t)(bm + (tid >> 2)) * lda + k0 + 32 + ((tid & 3) << 3)], 0, 1);
    }

    // ---- stage A tile: As[r][c] = f16(A[bm+r][k0+c]), 64x32 ----
    if constexpr (!GK) {
      // vectorized: 2 x (b128 load + pk cvt + ds_store_b64) per thread
#pragma unroll
      for (int j = 0; j < 2; ++j) {
        int id = tid + j * 256;              // 0..511 quads
        int r  = id >> 3;                    // 0..63
        int c4 = (id & 7) << 2;              // 0,4,...,28
        v4f v = *(const v4f*)&A[(size_t)(bm + r) * lda + k0 + c4];
        v4h h;
#pragma unroll
        for (int q = 0; q < 4; ++q) h[q] = (_Float16)v[q];
        *(v4h*)&As[r * 40 + c4] = h;
      }
    } else {
      // branchless clamp+select (only used when K ragged, e.g. K=24)
#pragma unroll
      for (int j = 0; j < 8; ++j) {
        int id = tid + j * 256;
        int r = id >> 5, c = id & 31;
        int k = k0 + c;
        int kc = k < K ? k : K - 1;
        float v = A[(size_t)(bm + r) * lda + kc];
        As[r * 40 + c] = (_Float16)(k < K ? v : 0.0f);
      }
    }

    // ---- stage B tile transposed into LDS: Bs[n][k] ----
    if (!wt) {
      // W is [K,N]: 4 coalesced loads down K per thread, one ds_store_b64
#pragma unroll
      for (int j = 0; j < 2; ++j) {
        int id = tid + j * 256;              // 0..511
        int n  = id & 63;
        int kq = (id >> 6) << 2;             // 0,4,...,28
        int gn = bn + n;
        int gnc = GN ? (gn < N ? gn : N - 1) : gn;
        v4h h;
#pragma unroll
        for (int q = 0; q < 4; ++q) {
          int gk = k0 + kq + q;
          int gkc = GK ? (gk < K ? gk : K - 1) : gk;
          float v = W[(size_t)gkc * ldw + gnc];
          bool ok = (!GK || gk < K) && (!GN || gn < N);
          h[q] = (_Float16)(ok ? v : 0.0f);
        }
        *(v4h*)&Bs[n * 40 + kq] = h;
      }
    } else {
      // W is [N,K]: vectorized along K (patch embed path; dims aligned)
#pragma unroll
      for (int j = 0; j < 2; ++j) {
        int id = tid + j * 256;
        int n  = id >> 3;
        int c4 = (id & 7) << 2;
        int gn = bn + n;
        if constexpr (!GK && !GN) {
          v4f v = *(const v4f*)&W[(size_t)gn * ldw + k0 + c4];
          v4h h;
#pragma unroll
          for (int q = 0; q < 4; ++q) h[q] = (_Float16)v[q];
          *(v4h*)&Bs[n * 40 + c4] = h;
        } else {
          int gnc = GN ? (gn < N ? gn : N - 1) : gn;
          v4h h;
#pragma unroll
          for (int q = 0; q < 4; ++q) {
            int gk = k0 + c4 + q;
            int gkc = GK ? (gk < K ? gk : K - 1) : gk;
            float v = W[(size_t)gnc * ldw + gkc];
            bool ok = (!GK || gk < K) && (!GN || gn < N);
            h[q] = (_Float16)(ok ? v : 0.0f);
          }
          *(v4h*)&Bs[n * 40 + c4] = h;
        }
      }
    }
    __syncthreads();

    // ---- fragments + WMMA ----
    {
      // A fragment (16x32 f16), per ISA layout
      int arow = wm * 16 + (lane & 15);
      int kb   = (lane >> 4) * 8;                       // 0 or 8
      v8h alo = *(const v8h*)&As[arow * 40 + kb];       // K = kb..kb+7
      v8h ahi = *(const v8h*)&As[arow * 40 + kb + 16];  // K = kb+16..kb+23
      v16h a;
#pragma unroll
      for (int i = 0; i < 8; ++i) { a[i] = alo[i]; a[i + 8] = ahi[i]; }

      // B fragments (32x16 f16): lane holds col N=lane&15, K=koff..koff+7, koff+16..koff+23
      int koff  = (lane >> 4) * 8;                      // 0 or 8
      int nidx0 = wn * 32 + (lane & 15);
      v8h b0lo = *(const v8h*)&Bs[nidx0 * 40 + koff];
      v8h b0hi = *(const v8h*)&Bs[nidx0 * 40 + koff + 16];
      v16h b0;
#pragma unroll
      for (int i = 0; i < 8; ++i) { b0[i] = b0lo[i]; b0[i + 8] = b0hi[i]; }
      int nidx1 = nidx0 + 16;
      v8h b1lo = *(const v8h*)&Bs[nidx1 * 40 + koff];
      v8h b1hi = *(const v8h*)&Bs[nidx1 * 40 + koff + 16];
      v16h b1;
#pragma unroll
      for (int i = 0; i < 8; ++i) { b1[i] = b1lo[i]; b1[i + 8] = b1hi[i]; }

      acc0 = __builtin_amdgcn_wmma_f32_16x16x32_f16(false, a, false, b0,
                                                    (short)0, acc0, false, false);
      acc1 = __builtin_amdgcn_wmma_f32_16x16x32_f16(false, a, false, b1,
                                                    (short)0, acc1, false, false);
    }
    __syncthreads();
  }

  // ---- epilogue: VGPR i -> M = i + 8*(lane>=16), N = lane&15 ----
#pragma unroll
  for (int i = 0; i < 8; ++i) {
    int m = mrow + i + ((lane >> 4) << 3);   // always < M (M%64==0)
#pragma unroll
    for (int s = 0; s < 2; ++s) {
      int n = ncol0 + s * 16 + (lane & 15);
      if (GN && n >= N) continue;
      float v = (s == 0) ? acc0[i] : acc1[i];
      float* cp = &C[(size_t)m * ldc + n];
      if (mode == 0) {
        *cp = bias ? v + bias[n] : v;
      } else if (mode == 1) {
        float x = v + bias[n];
        *cp = (x > 20.0f) ? x : log1pf(expf(x));
      } else if (mode == 2) {
        *cp += v;
      } else {
        *cp = v + bias[n] + pos[(size_t)(m % LSEQ) * DMODEL + n];
      }
    }
  }
}

// ---------------- im2col for patch embedding ----------------
__global__ void im2col_k(const float* __restrict__ img, float* __restrict__ out) {
  int idx = blockIdx.x * 256 + threadIdx.x;
  if (idx >= ROWS * 768) return;
  int col = idx % 768;
  int row = idx / 768;
  int b = row / LSEQ, l = row % LSEQ;
  int c  = col >> 8;          // /256
  int py = (col >> 4) & 15;
  int px = col & 15;
  int iy = (l / 14) * PATCH + py;
  int ix = (l % 14) * PATCH + px;
  out[idx] = img[((size_t)b * 3 + c) * IMG * IMG + (size_t)iy * IMG + ix];
}

// ---------------- LayerNorm: one wave per row ----------------
__global__ void layernorm_k(const float* __restrict__ x, const float* __restrict__ g,
                            const float* __restrict__ b, float* __restrict__ out,
                            int rows) {
  int row  = blockIdx.x * 8 + (threadIdx.x >> 5);
  int lane = threadIdx.x & 31;
  if (row >= rows) return;
  const float* xr = x + (size_t)row * DMODEL;
  float s = 0.0f, s2 = 0.0f;
  for (int i = lane; i < DMODEL; i += 32) {
    float v = xr[i];
    s += v; s2 += v * v;
  }
#pragma unroll
  for (int off = 16; off > 0; off >>= 1) {
    s  += __shfl_xor(s, off, 32);
    s2 += __shfl_xor(s2, off, 32);
  }
  float mu  = s / DMODEL;
  float var = s2 / DMODEL - mu * mu;
  float rs  = rsqrtf(var + 1e-5f);
  float* orow = out + (size_t)row * DMODEL;
  for (int i = lane; i < DMODEL; i += 32)
    orow[i] = (xr[i] - mu) * rs * g[i] + b[i];
}

// ---------------- depthwise causal conv (dir-aware) + SiLU ----------------
__global__ void conv_silu_k(const float* __restrict__ xz, const float* __restrict__ w,
                            const float* __restrict__ bias, float* __restrict__ out,
                            int rev) {
  int idx = blockIdx.x * 256 + threadIdx.x;
  if (idx >= ROWS * DINNER) return;
  int d   = idx % DINNER;
  int row = idx / DINNER;
  int b = row / LSEQ, l = row % LSEQ;
  float acc = bias[d];
#pragma unroll
  for (int j = 0; j < DCONV; ++j) {
    int ll = rev ? (l + (DCONV - 1) - j) : (l - (DCONV - 1) + j);
    if (ll >= 0 && ll < LSEQ)
      acc += w[d * DCONV + j] * xz[((size_t)(b * LSEQ + ll)) * (2 * DINNER) + d];
  }
  out[idx] = acc / (1.0f + expf(-acc));  // silu
}

// ---------------- selective scan, one thread per (batch, channel) --------------
// B/C (32 floats = 128B) for step t+1 are fetched with global_load_async_to_lds
// one step ahead (double-buffered), retired with s_wait_asynccnt 1, so the
// global-load latency is off the recurrence critical path.
__global__ __launch_bounds__(256) void scan_k(
    const float* __restrict__ DTp, const float* __restrict__ Up,
    const float* __restrict__ DBCp, const float* __restrict__ XZp,
    const float* __restrict__ Alog, const float* __restrict__ Dpv,
    float* __restrict__ Yp, int rev)
{
  __shared__ __align__(16) float sBC2[2][32];  // [buf][ B[0..15] | C[16..31] ]
  const int tid = threadIdx.x;
  const int b = blockIdx.x;
  const int d = blockIdx.y * 256 + tid;        // 0..767 over gridDim.y==3
  float An[DSTATE], h[DSTATE];
#pragma unroll
  for (int n = 0; n < DSTATE; ++n) {
    An[n] = -expf(Alog[(size_t)d * DSTATE + n]);
    h[n]  = 0.0f;
  }
  float Dpd = Dpv[d];

  // issue async B/C load for t = 0 into buffer 0 (lanes 0..7, 16B each)
  if (tid < 8) {
    int l0 = rev ? (LSEQ - 1) : 0;
    uint64_t g = (uint64_t)(uintptr_t)(DBCp + ((size_t)b * LSEQ + l0) * DBC_N + DTRANK + tid * 4);
    unsigned lo = (unsigned)(uintptr_t)&sBC2[0][tid * 4];
    asm volatile("global_load_async_to_lds_b128 %0, %1, off"
                 :: "v"(lo), "v"(g) : "memory");
  }

  for (int t = 0; t < LSEQ; ++t) {
    int buf = t & 1;
    int l = rev ? (LSEQ - 1 - t) : t;
    size_t row = (size_t)b * LSEQ + l;
    // issue next step's B/C into the other buffer (its readers finished at the
    // end-of-previous-iteration barrier)
    if (tid < 8 && t + 1 < LSEQ) {
      int l1 = rev ? (LSEQ - 2 - t) : (t + 1);
      uint64_t g = (uint64_t)(uintptr_t)(DBCp + ((size_t)b * LSEQ + l1) * DBC_N + DTRANK + tid * 4);
      unsigned lo = (unsigned)(uintptr_t)&sBC2[buf ^ 1][tid * 4];
      asm volatile("global_load_async_to_lds_b128 %0, %1, off"
                   :: "v"(lo), "v"(g) : "memory");
    }
    if (tid < 32) {  // wave 0 (the issuing wave): current buffer must be resident
      if (t + 1 < LSEQ) asm volatile("s_wait_asynccnt 0x1" ::: "memory");
      else              asm volatile("s_wait_asynccnt 0x0" ::: "memory");
    }
    __syncthreads();

    const float* sBC = sBC2[buf];
    float dt = DTp[row * DINNER + d];
    float u  = Up[row * DINNER + d];
    float z  = XZp[row * (2 * DINNER) + DINNER + d];
    float du = dt * u;
    float y  = 0.0f;
#pragma unroll
    for (int n = 0; n < DSTATE; ++n) {
      h[n] = expf(dt * An[n]) * h[n] + du * sBC[n];
      y   += h[n] * sBC[16 + n];
    }
    y += Dpd * u;
    y *= z / (1.0f + expf(-z));         // * silu(z)
    Yp[row * DINNER + d] = y;
    __syncthreads();                     // readers done before buf is re-filled
  }
}

// ---------------- mean pool over L ----------------
__global__ void pool_k(const float* __restrict__ xn, float* __restrict__ pooled) {
  int i = blockIdx.x * 256 + threadIdx.x;
  if (i >= NB * DMODEL) return;
  int b = i / DMODEL, dm = i % DMODEL;
  float s = 0.0f;
  for (int l = 0; l < LSEQ; ++l)
    s += xn[((size_t)b * LSEQ + l) * DMODEL + dm];
  pooled[i] = s * (1.0f / LSEQ);
}

// ---------------- heads (tiny) ----------------
__global__ void heads_k(const float* __restrict__ pooled,
                        const float* __restrict__ wd, const float* __restrict__ bd,
                        const float* __restrict__ wq, const float* __restrict__ bq,
                        float* __restrict__ out) {
  int i = blockIdx.x * 64 + threadIdx.x;
  if (i >= NB * (NDEV + NDIST)) return;
  if (i < NB * NDEV) {
    int b = i / NDEV, o = i % NDEV;
    float s = bd[o];
    for (int k = 0; k < DMODEL; ++k) s += pooled[b * DMODEL + k] * wd[k * NDEV + o];
    out[i] = s;
  } else {
    int j = i - NB * NDEV;
    int b = j / NDIST, o = j % NDIST;
    float s = bq[o];
    for (int k = 0; k < DMODEL; ++k) s += pooled[b * DMODEL + k] * wq[k * NDIST + o];
    out[i] = s;
  }
}

// ---------------- host orchestration ----------------
static inline dim3 gemm_grid(int M, int N) { return dim3((N + 63) / 64, (M + 63) / 64); }

extern "C" void kernel_launch(void* const* d_in, const int* in_sizes, int n_in,
                              void* d_out, int out_size, void* d_ws, size_t ws_size,
                              hipStream_t stream) {
  const float* images     = (const float*)d_in[0];
  const float* patch_w    = (const float*)d_in[1];
  const float* patch_b    = (const float*)d_in[2];
  const float* pos_emb    = (const float*)d_in[3];
  const float* ln_g       = (const float*)d_in[4];
  const float* ln_b       = (const float*)d_in[5];
  const float* in_w       = (const float*)d_in[6];
  const float* conv_w     = (const float*)d_in[7];
  const float* conv_b     = (const float*)d_in[8];
  const float* xproj_w    = (const float*)d_in[9];
  const float* dt_w       = (const float*)d_in[10];
  const float* dt_b       = (const float*)d_in[11];
  const float* A_log      = (const float*)d_in[12];
  const float* Dp         = (const float*)d_in[13];
  const float* out_w      = (const float*)d_in[14];
  const float* fln_g      = (const float*)d_in[15];
  const float* fln_b      = (const float*)d_in[16];
  const float* head_dev_w = (const float*)d_in[17];
  const float* head_dev_b = (const float*)d_in[18];
  const float* head_dist_w= (const float*)d_in[19];
  const float* head_dist_b= (const float*)d_in[20];

  float* ws = (float*)d_ws;
  float* X   = ws;                              // ROWS*DMODEL residual stream
  float* XN  = X   + (size_t)ROWS * DMODEL;     // ROWS*DMODEL
  float* XZ  = XN  + (size_t)ROWS * DMODEL;     // ROWS*2*DINNER
  float* UC  = XZ  + (size_t)ROWS * 2 * DINNER; // ROWS*DINNER (also im2col buf)
  float* DBC = UC  + (size_t)ROWS * DINNER;     // ROWS*56
  float* DTb = DBC + (size_t)ROWS * DBC_N;      // ROWS*DINNER
  float* Yb  = DTb + (size_t)ROWS * DINNER;     // ROWS*DINNER
  float* PO  = Yb  + (size_t)ROWS * DINNER;     // NB*DMODEL

  // ---- patch embed: im2col + WMMA GEMM (W transposed), epilogue +patch_b+pos ----
  im2col_k<<<(ROWS * 768 + 255) / 256, 256, 0, stream>>>(images, UC);
  gemm_wmma_f16<false, false><<<gemm_grid(ROWS, DMODEL), 256, 0, stream>>>(
      UC, 768, patch_w, 768, /*wt=*/1, patch_b, pos_emb,
      X, DMODEL, ROWS, DMODEL, 768, /*mode=*/3);

  for (int l = 0; l < NDEPTH; ++l) {
    layernorm_k<<<(ROWS + 7) / 8, 256, 0, stream>>>(
        X, ln_g + (size_t)l * DMODEL, ln_b + (size_t)l * DMODEL, XN, ROWS);
    for (int dir = 0; dir < 2; ++dir) {
      size_t w = (size_t)(l * 2 + dir);
      // in-proj: XZ = XN @ in_w[l,dir]   (6272 x 1536 x 384)
      gemm_wmma_f16<false, false><<<gemm_grid(ROWS, 2 * DINNER), 256, 0, stream>>>(
          XN, DMODEL, in_w + w * DMODEL * 2 * DINNER, 2 * DINNER, 0, nullptr, nullptr,
          XZ, 2 * DINNER, ROWS, 2 * DINNER, DMODEL, 0);
      // causal conv + silu (direction-aware)
      conv_silu_k<<<(ROWS * DINNER + 255) / 256, 256, 0, stream>>>(
          XZ, conv_w + w * DINNER * DCONV, conv_b + w * DINNER, UC, dir);
      // x-proj: DBC = UC @ xproj_w  (6272 x 56 x 768), N ragged
      gemm_wmma_f16<false, true><<<gemm_grid(ROWS, DBC_N), 256, 0, stream>>>(
          UC, DINNER, xproj_w + w * DINNER * DBC_N, DBC_N, 0, nullptr, nullptr,
          DBC, DBC_N, ROWS, DBC_N, DINNER, 0);
      // dt-proj + softplus: DTb = softplus(DBC[:, :24] @ dt_w + dt_b), K ragged
      gemm_wmma_f16<true, false><<<gemm_grid(ROWS, DINNER), 256, 0, stream>>>(
          DBC, DBC_N, dt_w + w * DTRANK * DINNER, DINNER, 0, dt_b + w * DINNER, nullptr,
          DTb, DINNER, ROWS, DINNER, DTRANK, 1);
      // selective scan + D skip + silu(z) gate (async B/C double-buffering)
      scan_k<<<dim3(NB, 3), 256, 0, stream>>>(
          DTb, UC, DBC, XZ, A_log + w * DINNER * DSTATE, Dp + w * DINNER, Yb, dir);
      // out-proj, accumulate into residual: X += Yb @ out_w
      gemm_wmma_f16<false, false><<<gemm_grid(ROWS, DMODEL), 256, 0, stream>>>(
          Yb, DINNER, out_w + w * DINNER * DMODEL, DMODEL, 0, nullptr, nullptr,
          X, DMODEL, ROWS, DMODEL, DINNER, 2);
    }
  }

  // ---- final LN -> mean pool -> heads ----
  layernorm_k<<<(ROWS + 7) / 8, 256, 0, stream>>>(X, fln_g, fln_b, XN, ROWS);
  pool_k<<<(NB * DMODEL + 255) / 256, 256, 0, stream>>>(XN, PO);
  heads_k<<<(NB * (NDEV + NDIST) + 63) / 64, 64, 0, stream>>>(
      PO, head_dev_w, head_dev_b, head_dist_w, head_dist_b, (float*)d_out);
}
